// MinGRULM_61984968016215
// MI455X (gfx1250) — matrix operand order repaired and checked by reading, not compiled, not run on hardware
//
#include <hip/hip_runtime.h>
#include <hip/hip_bf16.h>
#include <math.h>

// ---------------- problem constants ----------------
#define VV 50257
#define DD 1024
#define HH 2048
#define LL 6
#define KCONV 3
#define BB 4
#define SS 2048
#define MM (BB * SS)            // 8192 tokens
#define BSD ((size_t)MM * DD)   // 8388608
#define BSH ((size_t)MM * HH)   // 16777216
#define RMS_EPS 1.1920929e-07f

// scan chunking
#define NC 16
#define CHUNK (SS / NC)         // 128

typedef __attribute__((ext_vector_type(16))) _Float16 v16h;
typedef __attribute__((ext_vector_type(8)))  _Float16 v8h;
typedef __attribute__((ext_vector_type(4)))  _Float16 v4h;
typedef __attribute__((ext_vector_type(8)))  float    v8f;

// ====================================================================
// WMMA GEMM:  out[M,N] = A[M,K] * W[N,K]^T (+bias, epilogue variants)
// A, W in f16 row-major. M fixed = 8192 via grid. N,K multiples of 256/32.
// Block: 256 threads = 8 waves; wave tile 64x64; block tile 128x256.
// EPI: 0 = outF = acc+bias ; 1 = outF += acc+bias ; 2 = outH = f16(gelu(acc+bias))
// ====================================================================
template <int EPI>
__global__ __launch_bounds__(256)
void gemm_wmma_kernel(const _Float16* __restrict__ A,
                      const _Float16* __restrict__ W,
                      const float* __restrict__ bias,
                      float* __restrict__ outF,
                      _Float16* __restrict__ outH,
                      int N, int Kd)
{
    const int lane = threadIdx.x & 31;
    const int wave = threadIdx.x >> 5;
    const int m0 = blockIdx.y * 128 + (wave & 1) * 64;
    const int n0 = blockIdx.x * 256 + (wave >> 1) * 64;

    const int rsel = lane & 15;           // row-in-tile for both A and B frags
    const int ksA  = (lane >> 4) << 4;    // A: 0 or 16
    const int ksB  = (lane >> 4) << 4;    // B: 0 or 16

    v8f acc[4][4];
#pragma unroll
    for (int i = 0; i < 4; ++i)
#pragma unroll
        for (int j = 0; j < 4; ++j)
#pragma unroll
            for (int r = 0; r < 8; ++r)
                acc[i][j][r] = 0.0f;

    const _Float16* aBase = A + (size_t)(m0 + rsel) * Kd + ksA;
    const _Float16* bBase = W + (size_t)(n0 + rsel) * Kd + ksB;

    for (int kb = 0; kb < Kd; kb += 32) {
        v16h af[4], bf[4];
#pragma unroll
        for (int i = 0; i < 4; ++i) {
            // A fragment: rows m0+i*16+rsel ; elems 0..15 = K kb+ksA..kb+ksA+15
            const _Float16* pa = aBase + (size_t)(i * 16) * Kd + kb;
            v8h lo = *(const v8h*)pa;
            v8h hi = *(const v8h*)(pa + 8);
            af[i] = __builtin_shufflevector(lo, hi,
                     0,1,2,3,4,5,6,7,8,9,10,11,12,13,14,15);
            // B fragment: 16 K-contiguous halves of weight row n0+i*16+rsel
            const _Float16* pb = bBase + (size_t)(i * 16) * Kd + kb;
            bf[i] = *(const v16h*)pb;
        }
#pragma unroll
        for (int i = 0; i < 4; ++i)
#pragma unroll
            for (int j = 0; j < 4; ++j)
                acc[i][j] = __builtin_amdgcn_wmma_f32_16x16x32_f16(
                    false, af[i], false, bf[j], (short)0, acc[i][j],
                    false, false);
    }

    // epilogue: C/D layout -> M = r + (lane>=16)*8, N = lane&15
    const int rOff = (lane >> 4) << 3;
#pragma unroll
    for (int i = 0; i < 4; ++i) {
#pragma unroll
        for (int j = 0; j < 4; ++j) {
            const int col = n0 + j * 16 + (lane & 15);
            const float bv = bias[col];
#pragma unroll
            for (int r = 0; r < 8; ++r) {
                const int row = m0 + i * 16 + rOff + r;
                const size_t idx = (size_t)row * N + col;
                float v = acc[i][j][r] + bv;
                if (EPI == 0) {
                    outF[idx] = v;
                } else if (EPI == 1) {
                    outF[idx] += v;
                } else {
                    float g = 0.5f * v * (1.0f + erff(v * 0.70710678118654752f));
                    outH[idx] = (_Float16)g;
                }
            }
        }
    }
}

// ====================================================================
// elementwise / helper kernels
// ====================================================================
__global__ void embed_kernel(const int* __restrict__ ids,
                             const float* __restrict__ emb,
                             float* __restrict__ xe)
{
    size_t i = (size_t)blockIdx.x * 256 + threadIdx.x;   // over B*S*D
    int tok = ids[i >> 10];
    xe[i] = emb[(size_t)tok * DD + (i & 1023)];
}

__global__ void cvt16_kernel(const float* __restrict__ in,
                             _Float16* __restrict__ out)
{
    size_t i = (size_t)blockIdx.x * 256 + threadIdx.x;
    out[i] = (_Float16)in[i];
}

// depthwise conv K=3, 'SAME', per-channel weights w[d*3+k], writes f16
__global__ void dwconv_kernel(const float* __restrict__ xe,
                              const float* __restrict__ w,
                              const float* __restrict__ b,
                              _Float16* __restrict__ o)
{
    size_t i = (size_t)blockIdx.x * 256 + threadIdx.x;   // over B*S*D
    int d = (int)(i & 1023);
    size_t bs = i >> 10;
    int s = (int)(bs & (SS - 1));
    const float* xc = xe + bs * DD + d;
    float acc = b[d];
    float w0 = w[d * 3 + 0], w1 = w[d * 3 + 1], w2 = w[d * 3 + 2];
    if (s > 0)      acc += w0 * xc[-DD];
    acc += w1 * xc[0];
    if (s < SS - 1) acc += w2 * xc[DD];
    o[i] = (_Float16)acc;
}

// RMSNorm over D=1024, one block per token, writes f16 normalized*weight
__global__ __launch_bounds__(256)
void rmsnorm_kernel(const float* __restrict__ x,
                    const float* __restrict__ w,
                    _Float16* __restrict__ o)
{
    const int t = blockIdx.x;
    const int tid = threadIdx.x;
    const float4 v = ((const float4*)(x + (size_t)t * DD))[tid];
    __shared__ float red[256];
    red[tid] = v.x * v.x + v.y * v.y + v.z * v.z + v.w * v.w;
    __syncthreads();
    for (int off = 128; off > 0; off >>= 1) {
        if (tid < off) red[tid] += red[tid + off];
        __syncthreads();
    }
    const float scale = rsqrtf(red[0] * (1.0f / (float)DD) + RMS_EPS);
    const float4 wv = ((const float4*)w)[tid];
    v4h out;
    out[0] = (_Float16)(v.x * scale * wv.x);
    out[1] = (_Float16)(v.y * scale * wv.y);
    out[2] = (_Float16)(v.z * scale * wv.z);
    out[3] = (_Float16)(v.w * scale * wv.w);
    *(v4h*)(o + (size_t)t * DD + (size_t)tid * 4) = out;
}

// z = sigmoid(zpre); a = 1-z -> zpre ; bc = z*htil -> htil   (in place)
__global__ void combine_kernel(float* __restrict__ zpre,
                               float* __restrict__ htil)
{
    size_t i = (size_t)blockIdx.x * 256 + threadIdx.x;   // over B*S*H
    float z = 1.0f / (1.0f + expf(-zpre[i]));
    float ht = htil[i];
    zpre[i] = 1.0f - z;
    htil[i] = z * ht;
}

// ---- chunked scan: h_s = a_s*h_{s-1} + b_s over S, per (b,h) channel ----
__global__ void scan_pass1(const float* __restrict__ aB,
                           const float* __restrict__ bBuf,
                           float* __restrict__ sumA,
                           float* __restrict__ sumB)
{
    int tid = blockIdx.x * 256 + threadIdx.x;            // B*H*NC
    int chunk = tid & (NC - 1);
    int ch = tid >> 4;
    int b = ch >> 11;
    int hh = ch & (HH - 1);
    float A = 1.0f, Bc = 0.0f;
    int s0 = chunk * CHUNK;
    for (int s = s0; s < s0 + CHUNK; ++s) {
        size_t ix = ((size_t)(b * SS + s)) * HH + hh;
        float a = aB[ix], c = bBuf[ix];
        A = A * a;
        Bc = Bc * a + c;
    }
    sumA[tid] = A;
    sumB[tid] = Bc;
}

__global__ void scan_pass2(const float* __restrict__ sumA,
                           const float* __restrict__ sumB,
                           const float* __restrict__ hprev,
                           int l,
                           float* __restrict__ entry)
{
    int ch = blockIdx.x * 256 + threadIdx.x;             // B*H
    int b = ch >> 11;
    int hh = ch & (HH - 1);
    float e = hprev[((size_t)b * LL + l) * HH + hh];
    for (int c = 0; c < NC; ++c) {
        entry[(size_t)ch * NC + c] = e;
        e = sumA[(size_t)ch * NC + c] * e + sumB[(size_t)ch * NC + c];
    }
}

__global__ void scan_pass3(const float* __restrict__ aB,
                           const float* __restrict__ bBuf,
                           const float* __restrict__ entry,
                           _Float16* __restrict__ h16,
                           float* __restrict__ hnext,
                           int l)
{
    int tid = blockIdx.x * 256 + threadIdx.x;            // B*H*NC
    int chunk = tid & (NC - 1);
    int ch = tid >> 4;
    int b = ch >> 11;
    int hh = ch & (HH - 1);
    float h = entry[tid];
    int s0 = chunk * CHUNK;
    for (int s = s0; s < s0 + CHUNK; ++s) {
        size_t ix = ((size_t)(b * SS + s)) * HH + hh;
        h = aB[ix] * h + bBuf[ix];
        h16[ix] = (_Float16)h;
    }
    if (chunk == NC - 1)
        hnext[((size_t)b * LL + l) * HH + hh] = h;
}

// ====================================================================
// host-side orchestration
// ====================================================================
static inline void cvt_w(const float* src, _Float16* dst, size_t n, hipStream_t s)
{
    cvt16_kernel<<<(unsigned)(n / 256), 256, 0, s>>>(src, dst);
}

extern "C" void kernel_launch(void* const* d_in, const int* in_sizes, int n_in,
                              void* d_out, int out_size, void* d_ws, size_t ws_size,
                              hipStream_t stream)
{
    (void)in_sizes; (void)n_in; (void)out_size; (void)ws_size;

    const int*   ids       = (const int*)  d_in[0];
    const float* h_prev    = (const float*)d_in[1];
    const float* emb       = (const float*)d_in[2];
    const float* conv_dw_w = (const float*)d_in[3];
    const float* conv_dw_b = (const float*)d_in[4];
    const float* conv_pw_w = (const float*)d_in[5];
    const float* conv_pw_b = (const float*)d_in[6];
    const float* n1_w      = (const float*)d_in[7];
    const float* wz        = (const float*)d_in[8];
    const float* bz        = (const float*)d_in[9];
    const float* wh        = (const float*)d_in[10];
    const float* bh        = (const float*)d_in[11];
    const float* wo        = (const float*)d_in[12];
    const float* bo        = (const float*)d_in[13];
    const float* n2_w      = (const float*)d_in[14];
    const float* f1_w      = (const float*)d_in[15];
    const float* f1_b      = (const float*)d_in[16];
    const float* f2_w      = (const float*)d_in[17];
    const float* f2_b      = (const float*)d_in[18];
    const float* norm_w    = (const float*)d_in[19];
    const float* out_w     = (const float*)d_in[20];
    const float* out_b     = (const float*)d_in[21];

    float* out0  = (float*)d_out;                 // (B,S,H)
    float* hnext = out0 + BSH;                    // (B,L,H)

    // workspace carve (all sizes 256B-aligned)
    char* p = (char*)d_ws;
    float*    xe    = (float*)p;            p += BSD * 4;          // fp32 activation
    _Float16* act16 = (_Float16*)p;         p += BSD * 2;          // f16 GEMM input (B,S,D)
    float*    zpre  = (float*)p;            p += BSH * 4;          // z-pre / a
    float*    htil  = (float*)p;            p += BSH * 4;          // h~  / b
    _Float16* h16   = (_Float16*)p;         p += BSH * 2;          // h (f16) / gelu(f) (f16)
    _Float16* w16   = (_Float16*)p;         p += (size_t)HH * DD * 2;  // converted weights
    float*    sumA  = (float*)p;            p += (size_t)BB * HH * NC * 4;
    float*    sumB  = (float*)p;            p += (size_t)BB * HH * NC * 4;
    float*    entry = (float*)p;            p += (size_t)BB * HH * NC * 4;

    const dim3 blk(256);
    const unsigned gBSD = (unsigned)(BSD / 256);
    const unsigned gBSH = (unsigned)(BSH / 256);
    const unsigned gScan = (unsigned)((size_t)BB * HH * NC / 256);  // 512
    const unsigned gChan = (unsigned)((size_t)BB * HH / 256);       // 32
    const dim3 gemmGridD(DD / 256, MM / 128);   // N=1024
    const dim3 gemmGridH(HH / 256, MM / 128);   // N=2048

    // embedding gather
    embed_kernel<<<gBSD, blk, 0, stream>>>(ids, emb, xe);

    for (int l = 0; l < LL; ++l) {
        const float* dw_w = conv_dw_w + (size_t)l * DD * KCONV;
        const float* dw_b = conv_dw_b + (size_t)l * DD;
        const float* pw_w = conv_pw_w + (size_t)l * DD * DD;
        const float* pw_b = conv_pw_b + (size_t)l * DD;

        // depthwise conv -> f16
        dwconv_kernel<<<gBSD, blk, 0, stream>>>(xe, dw_w, dw_b, act16);
        // pointwise 1x1 conv GEMM, fused residual:  xe += c @ pw^T + pw_b
        cvt_w(pw_w, w16, (size_t)DD * DD, stream);
        gemm_wmma_kernel<1><<<gemmGridD, blk, 0, stream>>>(act16, w16, pw_b,
                                                           xe, nullptr, DD, DD);

        // RMSNorm -> f16
        rmsnorm_kernel<<<MM, blk, 0, stream>>>(xe, n1_w + (size_t)l * DD, act16);

        // z-pre and h~ GEMMs (D -> H)
        cvt_w(wz + (size_t)l * HH * DD, w16, (size_t)HH * DD, stream);
        gemm_wmma_kernel<0><<<gemmGridH, blk, 0, stream>>>(act16, w16,
            bz + (size_t)l * HH, zpre, nullptr, HH, DD);
        cvt_w(wh + (size_t)l * HH * DD, w16, (size_t)HH * DD, stream);
        gemm_wmma_kernel<0><<<gemmGridH, blk, 0, stream>>>(act16, w16,
            bh + (size_t)l * HH, htil, nullptr, HH, DD);

        // a = 1-sigmoid(z), b = sigmoid(z)*h~   (in place)
        combine_kernel<<<gBSH, blk, 0, stream>>>(zpre, htil);

        // chunked linear-recurrence scan; h -> f16, last step -> hnext
        scan_pass1<<<gScan, blk, 0, stream>>>(zpre, htil, sumA, sumB);
        scan_pass2<<<gChan, blk, 0, stream>>>(sumA, sumB, h_prev, l, entry);
        scan_pass3<<<gScan, blk, 0, stream>>>(zpre, htil, entry, h16, hnext, l);

        // output projection (H -> D), fused residual: xe += h @ wo^T + bo
        cvt_w(wo + (size_t)l * DD * HH, w16, (size_t)DD * HH, stream);
        gemm_wmma_kernel<1><<<gemmGridD, blk, 0, stream>>>(h16, w16,
            bo + (size_t)l * DD, xe, nullptr, DD, HH);

        // FFN: norm -> f1 (+gelu, f16 out) -> f2 (+residual)
        rmsnorm_kernel<<<MM, blk, 0, stream>>>(xe, n2_w + (size_t)l * DD, act16);
        cvt_w(f1_w + (size_t)l * HH * DD, w16, (size_t)HH * DD, stream);
        gemm_wmma_kernel<2><<<gemmGridH, blk, 0, stream>>>(act16, w16,
            f1_b + (size_t)l * HH, nullptr, h16, HH, DD);
        cvt_w(f2_w + (size_t)l * DD * HH, w16, (size_t)DD * HH, stream);
        gemm_wmma_kernel<1><<<gemmGridD, blk, 0, stream>>>(h16, w16,
            f2_b + (size_t)l * DD, xe, nullptr, DD, HH);
    }

    // final norm + head (D -> H) into d_out
    rmsnorm_kernel<<<MM, blk, 0, stream>>>(xe, norm_w, act16);
    cvt_w(out_w, w16, (size_t)HH * DD, stream);
    gemm_wmma_kernel<0><<<gemmGridH, blk, 0, stream>>>(act16, w16, out_b,
                                                       out0, nullptr, HH, DD);
}
